// AttentionHead_50268297232471
// MI455X (gfx1250) — matrix operand order repaired and checked by reading, not compiled, not run on hardware
//
#include <hip/hip_runtime.h>
#include <hip/hip_bf16.h>

// AttentionHead for MI455X (gfx1250, wave32, WMMA).
// B=4, S=2048, D_MODEL=1024, D_K=64.
// Phase 1: q/k/v projections via v_wmma_f32_16x16x32_f16 (f32 accum). One wave
//          computes a full 16x64 output tile: A tile loaded/converted once per
//          K-step and shared by 4 WMMAs. q scaled by D_MODEL^-0.5 = 1/32.
//          Outputs f16: q,k row-major [B*S,64]; v transposed [B,64,S].
// Phase 2: flash-attention per 16-query tile, online softmax in f32,
//          P transposed through per-wave LDS, PV via WMMA, mask prefetch.

#define AH_B  4
#define AH_S  2048
#define AH_DM 1024
#define AH_DK 64

typedef __attribute__((ext_vector_type(16))) _Float16 v16h;
typedef __attribute__((ext_vector_type(8)))  _Float16 v8h;
typedef __attribute__((ext_vector_type(8)))  float    v8f;
typedef __attribute__((ext_vector_type(4)))  float    v4f;

// ---------------------------------------------------------------------------
// Projection: out[s, d] = sum_m x[s, m] * w[d, m]    (x: [B*S,1024], w: [64,1024])
// Block = 128 threads (4 waves). Wave wv owns row-tile (blockIdx.x*4 + wv) and
// computes all 64 output columns: per 32-wide K-step, A is loaded+converted
// once and feeds 4 WMMAs (one per 16-col n-tile).
// transposed==0: store out as f16 [B*S, 64]   (for q, k)
// transposed==1: store out as f16 [B][64][S]  (for v; makes PV B-operand contiguous)
// ---------------------------------------------------------------------------
__global__ __launch_bounds__(128)
void ah_proj(const float* __restrict__ x, const float* __restrict__ w,
             _Float16* __restrict__ out, int transposed, float scale)
{
    const int lane = threadIdx.x & 31;
    const int wv   = threadIdx.x >> 5;
    const int hf   = lane >> 4;          // lane half
    const int l15  = lane & 15;
    const long row0 = ((long)blockIdx.x * 4 + wv) * 16;

    // A: row = row0 + l15; element e -> k = kk + hf*16 + e
    const float* arow = x + (row0 + l15) * AH_DM;
    // B (n-tile nt): ncol = nt*16 + l15; element e -> k = kk + hf*16 + e
    const float* brow = w + (long)l15 * AH_DM;

    v8f acc[4] = {{}, {}, {}, {}};
    #pragma unroll 2
    for (int kk = 0; kk < AH_DM; kk += 32) {
        v16h a;
        {
            const v4f* ap0 = (const v4f*)(arow + kk + hf * 16);
            const v4f* ap1 = (const v4f*)(arow + kk + hf * 16 + 8);
            v4f A0 = ap0[0], A1 = ap0[1], A2 = ap1[0], A3 = ap1[1];
            #pragma unroll
            for (int e = 0; e < 4; ++e) {
                a[e]      = (_Float16)A0[e];
                a[4 + e]  = (_Float16)A1[e];
                a[8 + e]  = (_Float16)A2[e];
                a[12 + e] = (_Float16)A3[e];
            }
        }
        #pragma unroll
        for (int nt = 0; nt < 4; ++nt) {
            v16h bm;
            const v4f* bp = (const v4f*)(brow + (long)nt * 16 * AH_DM + kk + hf * 16);
            v4f B0 = bp[0], B1 = bp[1], B2 = bp[2], B3 = bp[3];
            #pragma unroll
            for (int e = 0; e < 4; ++e) {
                bm[e]      = (_Float16)B0[e];
                bm[4 + e]  = (_Float16)B1[e];
                bm[8 + e]  = (_Float16)B2[e];
                bm[12 + e] = (_Float16)B3[e];
            }
            acc[nt] = __builtin_amdgcn_wmma_f32_16x16x32_f16(false, a, false, bm,
                                                             (short)0, acc[nt],
                                                             false, false);
        }
    }

    if (!transposed) {
        // C layout: row M = r + 8*hf, col N = l15 (within each 16-col n-tile)
        #pragma unroll
        for (int nt = 0; nt < 4; ++nt)
            #pragma unroll
            for (int r = 0; r < 8; ++r)
                out[(row0 + r + 8 * hf) * AH_DK + nt * 16 + l15] =
                    (_Float16)(acc[nt][r] * scale);
    } else {
        // vT[bidx][d][s]: for fixed lane (d fixed), s = sInB + hf*8 + r is
        // contiguous in r -> one 16-byte store per n-tile.
        const long bidx = row0 >> 11;       // row0 / S
        const long sInB = row0 & (AH_S - 1);
        #pragma unroll
        for (int nt = 0; nt < 4; ++nt) {
            const int d = nt * 16 + l15;
            v8h hv;
            #pragma unroll
            for (int r = 0; r < 8; ++r) hv[r] = (_Float16)(acc[nt][r] * scale);
            *(v8h*)(out + (bidx * AH_DK + d) * (long)AH_S + sInB + hf * 8) = hv;
        }
    }
}

// ---------------------------------------------------------------------------
// Flash attention. Grid: (S/16/4, B). Block = 128 threads; wave wv owns query
// tile (blockIdx.x*4 + wv). Streams 32-key tiles: 4 score WMMAs, online
// softmax (f32, shfl_xor over 16-lane halves), P -> LDS -> A-layout, 4 PV WMMAs.
// ---------------------------------------------------------------------------
__global__ __launch_bounds__(128)
void ah_attn(const _Float16* __restrict__ qh, const _Float16* __restrict__ kh,
             const _Float16* __restrict__ vT, const unsigned char* __restrict__ mask,
             float* __restrict__ out)
{
    const int lane = threadIdx.x & 31;
    const int wv   = threadIdx.x >> 5;
    const int hf   = lane >> 4;
    const int l15  = lane & 15;
    const int b    = blockIdx.y;
    const int qtile = blockIdx.x * 4 + wv;           // 0..127
    const long q0   = (long)b * AH_S + qtile * 16;   // global query row

    __shared__ __align__(16) _Float16 pstage[4][16 * 32];
    _Float16* pb = pstage[wv];

    // Load the q A-operands once: K = d_k = 64 -> two 16x32 A tiles.
    v16h aq[2];
    const _Float16* qrow = qh + (q0 + l15) * AH_DK;
    #pragma unroll
    for (int t = 0; t < 2; ++t) {
        v8h c0 = *(const v8h*)(qrow + t * 32 + hf * 16);
        v8h c1 = *(const v8h*)(qrow + t * 32 + hf * 16 + 8);
        #pragma unroll
        for (int e = 0; e < 8; ++e) { aq[t][e] = c0[e]; aq[t][8 + e] = c1[e]; }
    }

    float mrun[8], lrun[8];
    v8f acc[4] = {{}, {}, {}, {}};
    #pragma unroll
    for (int r = 0; r < 8; ++r) { mrun[r] = -3.0e38f; lrun[r] = 0.f; }

    const unsigned char* mbase = mask + ((long)b * AH_S + qtile * 16) * (long)AH_S;

    for (int key0 = 0; key0 < AH_S; key0 += 32) {
        // Prefetch the mask stream (only HBM-resident stream here) two key
        // tiles ahead; lane spread l15 covers all 16 query rows' cachelines.
        if (key0 + 64 < AH_S)
            __builtin_prefetch(mbase + (long)l15 * AH_S + key0 + 64, 0, 1);

        // ---- scores: two 16x16 tiles over keys [key0, key0+32) ----
        v8f sc[2];
        #pragma unroll
        for (int t = 0; t < 2; ++t) {
            const _Float16* krow = kh + ((long)b * AH_S + key0 + t * 16 + l15) * AH_DK;
            v8f cs = {};
            #pragma unroll
            for (int ks = 0; ks < 2; ++ks) {
                v16h bm;
                v8h b0 = *(const v8h*)(krow + ks * 32 + hf * 16);
                v8h b1 = *(const v8h*)(krow + ks * 32 + hf * 16 + 8);
                #pragma unroll
                for (int e = 0; e < 8; ++e) { bm[e] = b0[e]; bm[8 + e] = b1[e]; }
                cs = __builtin_amdgcn_wmma_f32_16x16x32_f16(false, aq[ks], false, bm,
                                                            (short)0, cs, false, false);
            }
            sc[t] = cs;
        }

        // ---- mask + online softmax (scale 1/32 already folded into q) ----
        #pragma unroll
        for (int r = 0; r < 8; ++r) {
            const int M = r + 8 * hf;
            const unsigned char* mp = mbase + (long)M * AH_S + key0 + l15;
            float s0 = mp[0]  ? 1e-9f : sc[0][r];
            float s1 = mp[16] ? 1e-9f : sc[1][r];
            float vmax = fmaxf(s0, s1);
            #pragma unroll
            for (int off = 1; off < 16; off <<= 1)
                vmax = fmaxf(vmax, __shfl_xor(vmax, off, 32));
            const float mo = mrun[r];
            const float mn = fmaxf(mo, vmax);
            const float corr = __expf(mo - mn);
            const float p0 = __expf(s0 - mn);
            const float p1 = __expf(s1 - mn);
            float psum = p0 + p1;
            #pragma unroll
            for (int off = 1; off < 16; off <<= 1)
                psum += __shfl_xor(psum, off, 32);
            lrun[r] = lrun[r] * corr + psum;
            mrun[r] = mn;
            #pragma unroll
            for (int dt = 0; dt < 4; ++dt) acc[dt][r] *= corr;
            // stage P row-major [M][32] for the transpose to A layout
            pb[M * 32 + l15]      = (_Float16)p0;
            pb[M * 32 + 16 + l15] = (_Float16)p1;
        }

        // cross-lane LDS visibility within the wave (DS ops are in-order per
        // wave, but be explicit about the RAW across lanes)
        asm volatile("s_wait_dscnt 0x0" ::: "memory");

        // ---- P as A-operand (16 q-rows x 32 keys) ----
        v16h ap;
        {
            v8h c0 = *(const v8h*)(pb + l15 * 32 + hf * 16);
            v8h c1 = *(const v8h*)(pb + l15 * 32 + hf * 16 + 8);
            #pragma unroll
            for (int e = 0; e < 8; ++e) { ap[e] = c0[e]; ap[8 + e] = c1[e]; }
        }

        // ---- PV: acc[dt] += P(16x32) x V(32x16) per 16-wide d tile ----
        #pragma unroll
        for (int dt = 0; dt < 4; ++dt) {
            const _Float16* vrow = vT + ((long)b * AH_DK + dt * 16 + l15) * (long)AH_S
                                      + key0 + hf * 16;
            v16h bv;
            v8h b0 = *(const v8h*)(vrow);
            v8h b1 = *(const v8h*)(vrow + 8);
            #pragma unroll
            for (int e = 0; e < 8; ++e) { bv[e] = b0[e]; bv[8 + e] = b1[e]; }
            acc[dt] = __builtin_amdgcn_wmma_f32_16x16x32_f16(false, ap, false, bv,
                                                             (short)0, acc[dt], false, false);
        }
    }

    // ---- normalize and store f32 output [B,S,64] ----
    #pragma unroll
    for (int dt = 0; dt < 4; ++dt) {
        #pragma unroll
        for (int r = 0; r < 8; ++r) {
            const int M = r + 8 * hf;
            out[(q0 + M) * AH_DK + dt * 16 + l15] = acc[dt][r] / lrun[r];
        }
    }
}

// ---------------------------------------------------------------------------
extern "C" void kernel_launch(void* const* d_in, const int* in_sizes, int n_in,
                              void* d_out, int out_size, void* d_ws, size_t ws_size,
                              hipStream_t stream)
{
    const float* query = (const float*)d_in[0];
    const float* key   = (const float*)d_in[1];
    const float* value = (const float*)d_in[2];
    const unsigned char* mask = (const unsigned char*)d_in[3]; // jnp.bool_ = 1 byte
    const float* w_q = (const float*)d_in[4];
    const float* w_k = (const float*)d_in[5];
    const float* w_v = (const float*)d_in[6];
    float* out = (float*)d_out;

    // Workspace: qh, kh [B*S,64] f16; vT [B,64,S] f16  => 3 MiB total.
    _Float16* qh = (_Float16*)d_ws;
    _Float16* kh = qh + (size_t)AH_B * AH_S * AH_DK;
    _Float16* vT = kh + (size_t)AH_B * AH_S * AH_DK;

    dim3 pgrid(AH_B * AH_S / 64);   // 128 blocks x 4 waves = one wave per 16-row tile
    // q gets the D_MODEL^-0.5 = 1/32 score scale folded in (applied pre-mask,
    // matching the reference which scales scores before the mask fill).
    ah_proj<<<pgrid, 128, 0, stream>>>(query, w_q, qh, 0, 0.03125f);
    ah_proj<<<pgrid, 128, 0, stream>>>(key,   w_k, kh, 0, 1.0f);
    ah_proj<<<pgrid, 128, 0, stream>>>(value, w_v, vT, 1, 1.0f);

    dim3 agrid(AH_S / 16 / 4, AH_B); // (32, 4), 128 blocks x 4 waves = 512 waves of work
    ah_attn<<<agrid, 128, 0, stream>>>(qh, kh, vT, mask, out);
}
